// GDN_29961691857783
// MI455X (gfx1250) — matrix operand order repaired and checked by reading, not compiled, not run on hardware
//
#include <hip/hip_runtime.h>
#include <math.h>

// ---------------------------------------------------------------------------
// GDN forward for MI455X (gfx1250, wave32, WMMA).
// Bandwidth-bound (~300MB traffic @23.3TB/s ~ 13us floor, ~3 GFLOP compute).
// GEMMs use v_wmma_f32_16x16x32_f16 (f16 A/B, f32 acc); each wave computes a
// 16x64 C strip (4 accumulators) so the A fragment is reused across 4 WMMAs.
// ---------------------------------------------------------------------------

typedef __attribute__((ext_vector_type(16))) _Float16 v16h;
typedef __attribute__((ext_vector_type(8)))  float    v8f;

#define BB   128
#define NN   512
#define DD   64
#define II   256
#define BN   65536
#define KTOP 16
#define EPSF 1e-5f

// ---------------- graph build ------------------------------------------------
__global__ __launch_bounds__(256) void node_norms(const float* __restrict__ emb,
                                                  float* __restrict__ nrm) {
  int n = blockIdx.x * blockDim.x + threadIdx.x;
  if (n < NN) {
    float s = 0.f;
    for (int d = 0; d < DD; ++d) { float v = emb[n * DD + d]; s += v * v; }
    nrm[n] = sqrtf(s);
  }
}

__global__ __launch_bounds__(256) void build_graph(const float* __restrict__ emb,
                                                   const float* __restrict__ nrm,
                                                   int* __restrict__ nbr) {
  __shared__ float embi[DD];
  __shared__ float cosv[NN];
  __shared__ float rv[256];
  __shared__ int   ri[256];
  const int i = blockIdx.x;
  const int t = threadIdx.x;
  if (t < DD) embi[t] = emb[i * DD + t];
  __syncthreads();
  const float inv_ni = 1.0f / nrm[i];
  for (int j = t; j < NN; j += 256) {
    float dot = 0.f;
    for (int d = 0; d < DD; ++d) dot += embi[d] * emb[j * DD + d];
    cosv[j] = dot * inv_ni / nrm[j];
  }
  __syncthreads();
  for (int k = 0; k < KTOP; ++k) {
    float bv = -__builtin_inff(); int bi = 0x7fffffff;
    for (int j = t; j < NN; j += 256) {
      float v = cosv[j];
      if (v > bv || (v == bv && j < bi)) { bv = v; bi = j; }
    }
    rv[t] = bv; ri[t] = bi;
    __syncthreads();
    for (int s = 128; s > 0; s >>= 1) {
      if (t < s) {
        float ov = rv[t + s]; int oi = ri[t + s];
        if (ov > rv[t] || (ov == rv[t] && oi < ri[t])) { rv[t] = ov; ri[t] = oi; }
      }
      __syncthreads();
    }
    if (t == 0) { nbr[i * KTOP + k] = ri[0]; cosv[ri[0]] = -__builtin_inff(); }
    __syncthreads();
  }
}

// ---------------- WMMA GEMM: C[M,N] = A[M,64] * W[N,64]^T (+bias) ------------
// K fixed at 64 (both GEMMs) -> k-loop fully unrolled, no branch around WMMA.
// 8 waves / block; each wave owns a 16x(16*NT) C strip: A fragment loaded once
// per k-step, reused by NT WMMAs (4x fewer A fetches, 4x WMMA density).
template <int NT>
__global__ __launch_bounds__(256) void gemm_wmma(const float* __restrict__ A,
                                                 const float* __restrict__ W,
                                                 const float* __restrict__ bias,
                                                 float* __restrict__ C,
                                                 int M, int N) {
  const int wave = threadIdx.x >> 5;
  const int lane = threadIdx.x & 31;
  const int lo   = lane & 15;        // M row (A) / N col (B,C)
  const int hiH  = lane >> 4;        // half-wave selector
  const int tile_m = (blockIdx.x * 8 + wave) * 16;
  const int tile_n = blockIdx.y * (16 * NT);
  if (tile_m >= M) return;           // wave-uniform; EXEC stays all-1s for WMMA

  v8f acc[NT];
#pragma unroll
  for (int nt = 0; nt < NT; ++nt) acc[nt] = (v8f){};

  const float* __restrict__ arow = A + (size_t)(tile_m + lo) * 64;
  const float* __restrict__ wrow = W + (size_t)(tile_n + lo) * 64;

#pragma unroll
  for (int k0 = 0; k0 < 64; k0 += 32) {
    // A 16x32 f16: lane<16 -> K 0..15; lane>=16 -> K 16..31
    const int ab = k0 + hiH * 16;
    v16h a;
#pragma unroll
    for (int t = 0; t < 8; ++t) {
      a[t]     = (_Float16)arow[ab + t];
      a[8 + t] = (_Float16)arow[ab + 8 + t];
    }
    // B 32x16 f16 (fed transposed from W[N,64]): lane<16 -> K 0..15; >=16 -> 16..31
    const int bb = k0 + hiH * 16;
#pragma unroll
    for (int nt = 0; nt < NT; ++nt) {
      v16h b;
#pragma unroll
      for (int t = 0; t < 16; ++t)
        b[t] = (_Float16)wrow[(size_t)nt * 16 * 64 + bb + t];
      acc[nt] = __builtin_amdgcn_wmma_f32_16x16x32_f16(false, a, false, b,
                                                       (short)0, acc[nt],
                                                       false, false);
    }
  }

#pragma unroll
  for (int nt = 0; nt < NT; ++nt) {
    const int col = tile_n + nt * 16 + lo;
    const float bv = bias ? bias[col] : 0.f;
#pragma unroll
    for (int r = 0; r < 8; ++r) {
      const int row = tile_m + r + hiH * 8;
      C[(size_t)row * N + col] = acc[nt][r] + bv;
    }
  }
}

// ---------------- per-node attention scalars ---------------------------------
__global__ __launch_bounds__(256) void att_scalars(const float* __restrict__ h,
    const float* __restrict__ emb,
    const float* __restrict__ ai, const float* __restrict__ aj,
    const float* __restrict__ aei, const float* __restrict__ aej,
    float* __restrict__ hi, float* __restrict__ hj,
    float* __restrict__ ei, float* __restrict__ ej) {
  const int r = blockIdx.x * blockDim.x + threadIdx.x;
  if (r < BN) {
    float si = 0.f, sj = 0.f;
    for (int d = 0; d < DD; ++d) { float v = h[(size_t)r * DD + d]; si += v * ai[d]; sj += v * aj[d]; }
    hi[r] = si; hj[r] = sj;
  }
  if (r < NN) {
    float si = 0.f, sj = 0.f;
    for (int d = 0; d < DD; ++d) { float v = emb[r * DD + d]; si += v * aei[d]; sj += v * aej[d]; }
    ei[r] = si; ej[r] = sj;
  }
}

// ---------------- attention softmax + neighbor aggregation -------------------
__global__ __launch_bounds__(256) void aggregate(const float* __restrict__ h,
    const int* __restrict__ nbr,
    const float* __restrict__ hi, const float* __restrict__ hj,
    const float* __restrict__ ei, const float* __restrict__ ej,
    const float* __restrict__ gbias, float* __restrict__ out) {
  __shared__ float a_s[4][KTOP];
  __shared__ int   s_s[4][KTOP];
  const int ty = threadIdx.y, tx = threadIdx.x;   // blockDim = (64,4)
  const int r = blockIdx.x * 4 + ty;              // global row (b*N + i)
  const int i = r & (NN - 1);
  const int b = r >> 9;
  if (tx < KTOP) {
    const int src = nbr[i * KTOP + tx];
    const int sg  = b * NN + src;
    float a = hi[r] + ei[i] + hj[sg] + ej[src];
    a = (a > 0.f) ? a : 0.2f * a;                 // leaky_relu(0.2)
    a_s[ty][tx] = a;
    s_s[ty][tx] = sg;
  }
  __syncthreads();
  float amax = -__builtin_inff();
  for (int k = 0; k < KTOP; ++k) amax = fmaxf(amax, a_s[ty][k]);
  float den = 0.f;
  for (int k = 0; k < KTOP; ++k) den += __expf(a_s[ty][k] - amax);
  const float rden = 1.0f / den;
  float acc = 0.f;
  for (int k = 0; k < KTOP; ++k) {
    const float al = __expf(a_s[ty][k] - amax) * rden;
    acc += al * h[(size_t)s_s[ty][k] * DD + tx];
  }
  out[(size_t)r * DD + tx] = acc + gbias[tx];
}

// ---------------- utility: zero a float buffer -------------------------------
__global__ void zero_buf(float* __restrict__ p, int n) {
  int i = blockIdx.x * blockDim.x + threadIdx.x;
  if (i < n) p[i] = 0.f;
}

// ---------------- per-channel sum / sumsq (C == blockDim.x) ------------------
__global__ __launch_bounds__(256) void ch_stats(const float* __restrict__ a, int rows,
                                                float* __restrict__ sum,
                                                float* __restrict__ sq) {
  const int C = blockDim.x;
  const int c = threadIdx.x, ty = threadIdx.y;
  float s = 0.f, q = 0.f;
  for (int r = blockIdx.x * blockDim.y + ty; r < rows; r += gridDim.x * blockDim.y) {
    float v = a[(size_t)r * C + c];
    s += v; q += v * v;
  }
  __shared__ float ls[256], lq[256];
  const int idx = ty * C + c;
  ls[idx] = s; lq[idx] = q;
  __syncthreads();
  for (int st = blockDim.y >> 1; st > 0; st >>= 1) {
    if (ty < st) { ls[idx] += ls[idx + st * C]; lq[idx] += lq[idx + st * C]; }
    __syncthreads();
  }
  if (ty == 0) { atomicAdd(&sum[c], ls[c]); atomicAdd(&sq[c], lq[c]); }
}

// ---------------- BN1 apply + ReLU + emb-mul, fused BN2 stats ---------------
__global__ __launch_bounds__(256) void bn1_apply(const float* __restrict__ t1,
    const float* __restrict__ emb,
    const float* __restrict__ g, const float* __restrict__ bp,
    const float* __restrict__ sum1, const float* __restrict__ sq1,
    float* __restrict__ t2, float* __restrict__ sum2, float* __restrict__ sq2) {
  const int c = threadIdx.x;                      // 64 channels
  const int ty = threadIdx.y;                     // 4
  const float inv = 1.0f / (float)BN;
  const float m   = sum1[c] * inv;
  const float var = sq1[c] * inv - m * m;
  const float rs  = rsqrtf(var + EPSF);
  const float gg = g[c], bb = bp[c];
  float s = 0.f, q = 0.f;
  for (int r = blockIdx.x * 4 + ty; r < BN; r += gridDim.x * 4) {
    float v = (t1[(size_t)r * DD + c] - m) * rs * gg + bb;
    v = fmaxf(v, 0.f);
    v *= emb[(r & (NN - 1)) * DD + c];
    t2[(size_t)r * DD + c] = v;
    s += v; q += v * v;
  }
  __shared__ float ls[256], lq[256];
  const int idx = ty * DD + c;
  ls[idx] = s; lq[idx] = q;
  __syncthreads();
  for (int st = 2; st > 0; st >>= 1) {
    if (ty < st) { ls[idx] += ls[idx + st * DD]; lq[idx] += lq[idx + st * DD]; }
    __syncthreads();
  }
  if (ty == 0) { atomicAdd(&sum2[c], ls[c]); atomicAdd(&sq2[c], lq[c]); }
}

// ---------------- BN2 apply + ReLU ------------------------------------------
__global__ __launch_bounds__(256) void bn2_apply(const float* __restrict__ t2,
    const float* __restrict__ g, const float* __restrict__ bp,
    const float* __restrict__ sum2, const float* __restrict__ sq2,
    float* __restrict__ y) {
  const int idx = blockIdx.x * 256 + threadIdx.x;
  if (idx >= BN * DD) return;
  const int c = idx & (DD - 1);
  const float inv = 1.0f / (float)BN;
  const float m   = sum2[c] * inv;
  const float var = sq2[c] * inv - m * m;
  const float rs  = rsqrtf(var + EPSF);
  float v = (t2[idx] - m) * rs * g[c] + bp[c];
  y[idx] = fmaxf(v, 0.f);
}

// ---------------- BN3 apply + ReLU + dot(w2) + b2 ---------------------------
__global__ __launch_bounds__(256) void final_out(const float* __restrict__ z,
    const float* __restrict__ g, const float* __restrict__ bp,
    const float* __restrict__ sum3, const float* __restrict__ sq3,
    const float* __restrict__ w2, const float* __restrict__ b2,
    float* __restrict__ out) {
  const int c = threadIdx.x;                      // 256 channels
  const float inv = 1.0f / (float)BN;
  const float m   = sum3[c] * inv;
  const float var = sq3[c] * inv - m * m;
  const float gg  = g[c] * rsqrtf(var + EPSF);
  const float bb  = bp[c];
  const float w   = w2[c];
  __shared__ float red[256];
  for (int r = blockIdx.x; r < BN; r += gridDim.x) {
    float v = (z[(size_t)r * II + c] - m) * gg + bb;
    v = fmaxf(v, 0.f) * w;
    red[c] = v;
    __syncthreads();
    for (int st = 128; st > 0; st >>= 1) {
      if (c < st) red[c] += red[c + st];
      __syncthreads();
    }
    if (c == 0) out[r] = red[0] + b2[0];
    __syncthreads();
  }
}

// ---------------------------------------------------------------------------
extern "C" void kernel_launch(void* const* d_in, const int* in_sizes, int n_in,
                              void* d_out, int out_size, void* d_ws, size_t ws_size,
                              hipStream_t stream) {
  (void)in_sizes; (void)n_in; (void)out_size; (void)ws_size;
  const float* x       = (const float*)d_in[0];
  const float* emb     = (const float*)d_in[1];
  const float* lin_w   = (const float*)d_in[2];
  const float* att_i   = (const float*)d_in[3];
  const float* att_j   = (const float*)d_in[4];
  const float* att_em_i= (const float*)d_in[5];
  const float* att_em_j= (const float*)d_in[6];
  const float* gnn_b   = (const float*)d_in[7];
  const float* bn1_g   = (const float*)d_in[8];
  const float* bn1_b   = (const float*)d_in[9];
  const float* bn2_g   = (const float*)d_in[10];
  const float* bn2_b   = (const float*)d_in[11];
  const float* w1      = (const float*)d_in[12];
  const float* b1      = (const float*)d_in[13];
  const float* bn3_g   = (const float*)d_in[14];
  const float* bn3_b   = (const float*)d_in[15];
  const float* w2      = (const float*)d_in[16];
  const float* b2      = (const float*)d_in[17];
  float* out = (float*)d_out;

  // workspace carve-up (256B aligned), ~101 MB total; buffers reused:
  //   bufA: h  -> t2       bufB: t1 -> y        bufZ: z
  char* ws = (char*)d_ws;
  size_t off = 0;
  auto carve = [&](size_t bytes) -> void* {
    void* p = ws + off;
    off = (off + bytes + 255) & ~(size_t)255;
    return p;
  };
  int*   nbr  = (int*)  carve(NN * KTOP * 4);
  float* nrm  = (float*)carve(NN * 4);
  float* hi   = (float*)carve((size_t)BN * 4);
  float* hj   = (float*)carve((size_t)BN * 4);
  float* ei   = (float*)carve(NN * 4);
  float* ej   = (float*)carve(NN * 4);
  float* stats= (float*)carve(768 * 4);
  float* sum1 = stats,       *sq1 = stats + 64;
  float* sum2 = stats + 128, *sq2 = stats + 192;
  float* sum3 = stats + 256, *sq3 = stats + 512;
  float* bufA = (float*)carve((size_t)BN * DD * 4);   // h, then t2
  float* bufB = (float*)carve((size_t)BN * DD * 4);   // t1, then y
  float* bufZ = (float*)carve((size_t)BN * II * 4);   // z

  // stats must start at zero each call (atomic accumulation)
  zero_buf<<<3, 256, 0, stream>>>(stats, 768);

  // graph build
  node_norms<<<2, 256, 0, stream>>>(emb, nrm);
  build_graph<<<NN, 256, 0, stream>>>(emb, nrm, nbr);

  // GEMM1: h = X @ lin_w^T   [65536,64] = [65536,64]x[64,64]^T
  // one wave covers the full N=64 extent (NT=4)
  gemm_wmma<4><<<dim3(BN / 128, 1), 256, 0, stream>>>(
      x, lin_w, nullptr, bufA, BN, DD);

  // per-node attention scalars
  att_scalars<<<BN / 256, 256, 0, stream>>>(bufA, emb, att_i, att_j,
                                            att_em_i, att_em_j, hi, hj, ei, ej);

  // softmax-weighted neighbor aggregation + gnn_bias -> t1
  aggregate<<<BN / 4, dim3(64, 4), 0, stream>>>(bufA, nbr, hi, hj, ei, ej,
                                                gnn_b, bufB);

  // BN1 stats, then apply BN1+ReLU+emb-mul (fused BN2 stats) -> t2
  ch_stats<<<dim3(128), dim3(64, 4), 0, stream>>>(bufB, BN, sum1, sq1);
  bn1_apply<<<128, dim3(64, 4), 0, stream>>>(bufB, emb, bn1_g, bn1_b,
                                             sum1, sq1, bufA, sum2, sq2);

  // BN2 apply + ReLU -> y
  bn2_apply<<<(BN * DD) / 256, 256, 0, stream>>>(bufA, bn2_g, bn2_b,
                                                 sum2, sq2, bufB);

  // GEMM2: z = Y @ w1^T + b1   [65536,256], 16x64 strip per wave (NT=4)
  gemm_wmma<4><<<dim3(BN / 128, II / 64), 256, 0, stream>>>(
      bufB, w1, b1, bufZ, BN, II);

  // BN3 stats, then BN3+ReLU+dot(w2)+b2 -> out
  ch_stats<<<dim3(256), dim3(256, 1), 0, stream>>>(bufZ, BN, sum3, sq3);
  final_out<<<4096, 256, 0, stream>>>(bufZ, bn3_g, bn3_b, sum3, sq3, w2, b2, out);
}
